// MultiHeadAttention_12979391168606
// MI455X (gfx1250) — matrix operand order repaired and checked by reading, not compiled, not run on hardware
//
#include <hip/hip_runtime.h>
#include <hip/hip_bf16.h>

typedef __attribute__((ext_vector_type(16))) _Float16 v16h;
typedef __attribute__((ext_vector_type(8)))  _Float16 v8h;
typedef __attribute__((ext_vector_type(8)))  float    v8f;
typedef __attribute__((ext_vector_type(4)))  unsigned int u32x4;
typedef __attribute__((ext_vector_type(8)))  unsigned int u32x8;

#define LOG2E 1.4426950408889634f

// ---------------- WMMA helpers (CDNA5 16x16x32 f16 -> f32) ----------------

static __device__ __forceinline__ v8f wmma16(v16h a, v16h b, v8f c) {
  return __builtin_amdgcn_wmma_f32_16x16x32_f16(false, a, false, b, (short)0, c,
                                                false, false);
}

static __device__ __forceinline__ v16h cat8(v8h lo, v8h hi) {
  v16h r;
#pragma unroll
  for (int i = 0; i < 8; ++i) { r[i] = lo[i]; r[i + 8] = hi[i]; }
  return r;
}

// A fragment (16x32, row-major source). lane<16: row=lane, K={k0..+15};
// lane>=16: row=lane-16, K offset +16.
static __device__ __forceinline__ v16h load_afrag(const _Float16* A, int lda,
                                                  int k0, int l16, int g) {
  const _Float16* p = A + (size_t)l16 * lda + k0 + g * 16;
  return cat8(*(const v8h*)p, *(const v8h*)(p + 8));
}

// ---------------- Tensor Data Mover: 2D tile (f16) global -> LDS ----------------
// D# per CDNA5 ISA.  2 SGPR groups (<=2D tensor).

static __device__ __forceinline__ void tdm_load_2d_f16(const void* gaddr,
                                                       unsigned lds_off,
                                                       unsigned tile_d0,
                                                       unsigned tile_d1,
                                                       unsigned tensor_d0,
                                                       unsigned tensor_d1,
                                                       unsigned stride_d0) {
  unsigned long long ga = (unsigned long long)gaddr;
  u32x4 g0;
  g0[0] = 1u;                                    // count=1, user mode
  g0[1] = lds_off;                               // lds_addr (bytes)
  g0[2] = (unsigned)ga;                          // global_addr[31:0]
  g0[3] = (unsigned)(ga >> 32) | (2u << 30);     // global_addr[56:32] | type=2
  u32x8 g1;
  g1[0] = 1u << 16;                              // data_size=1 (2 bytes)
  g1[1] = (tensor_d0 & 0xFFFFu) << 16;           // tensor_dim0[15:0]
  g1[2] = (tensor_d0 >> 16) | ((tensor_d1 & 0xFFFFu) << 16);
  g1[3] = (tensor_d1 >> 16) | (tile_d0 << 16);   // tile_dim0
  g1[4] = tile_d1;                               // tile_dim1 (tile_dim2 = 0)
  g1[5] = stride_d0;                             // tensor_dim0_stride[31:0]
  g1[6] = 0u;
  g1[7] = 0u;
  asm volatile("tensor_load_to_lds %0, %1" :: "s"(g0), "s"(g1) : "memory");
}

// ---------------- conversion / transpose kernels ----------------

__global__ void k_cvt_f16(const float* __restrict__ in, _Float16* __restrict__ out, int n) {
  int i = blockIdx.x * blockDim.x + threadIdx.x;
  if (i < n) out[i] = (_Float16)in[i];
}

// w[H=16][D=1024][Dh=64] f32  ->  wT[H][Dh][D] f16 (scaled)
__global__ void k_cvt_wT(const float* __restrict__ w, _Float16* __restrict__ wT, float scale) {
  int i = blockIdx.x * blockDim.x + threadIdx.x;   // 16*64*1024 threads
  int d = i & 1023, e = (i >> 10) & 63, h = i >> 16;
  wT[i] = (_Float16)(w[((size_t)h << 16) + (size_t)d * 64 + e] * scale);
}

// w_o[K=1024][N=1024] f32 -> woT[N][K] f16
__global__ void k_cvt_woT(const float* __restrict__ w, _Float16* __restrict__ wT) {
  int i = blockIdx.x * blockDim.x + threadIdx.x;   // 1024*1024 threads
  int k = i & 1023, n = i >> 10;
  wT[i] = (_Float16)w[(size_t)k * 1024 + n];
}

// ---------------- QKV projection, 32x64 wave tile, double-buffered TDM ----------------
// grid (32, 48): y = matrix (0..15 Q, 16..31 K, 32..47 V); x*8+wave = 32-row tile.

__global__ __launch_bounds__(256) void k_qkv(const _Float16* __restrict__ E,
                                             const _Float16* __restrict__ wqT,
                                             const _Float16* __restrict__ wkT,
                                             const _Float16* __restrict__ wvT,
                                             _Float16* __restrict__ Q,
                                             _Float16* __restrict__ Kh,
                                             _Float16* __restrict__ Vt) {
  __shared__ _Float16 wlds[2][64 * 128];   // 2 x 16 KB weight panels
  const int wid = threadIdx.x >> 5, lane = threadIdx.x & 31;
  const int l16 = lane & 15, g = lane >> 4;
  const int mi = blockIdx.y, which = mi / 16, h = mi % 16;
  const int m0 = (blockIdx.x * 8 + wid) * 32;
  const _Float16* Wt = (which == 0 ? wqT : which == 1 ? wkT : wvT) + ((size_t)h << 16);
  const _Float16* A = E + (size_t)m0 * 1024;
  const unsigned lds0 = (unsigned)(size_t)(void*)&wlds[0][0];
  const unsigned lds1 = (unsigned)(size_t)(void*)&wlds[1][0];

  v8f acc[2][4] = {};
  if (wid == 0)
    tdm_load_2d_f16(Wt, lds0, 128, 64, 1024, 64, 1024);   // prime panel 0
  for (int c = 0; c < 8; ++c) {
    if (wid == 0) {
      if (c + 1 < 8) {   // stream next panel while this one is consumed
        tdm_load_2d_f16(Wt + (c + 1) * 128, ((c + 1) & 1) ? lds1 : lds0,
                        128, 64, 1024, 64, 1024);
        __builtin_amdgcn_s_wait_tensorcnt(1);   // in-order: panel c resident
      } else {
        __builtin_amdgcn_s_wait_tensorcnt(0);
      }
    }
    __syncthreads();                    // panel c visible to all waves
    const _Float16* wp = &wlds[c & 1][0];
    const int k0 = c * 128;
#pragma unroll
    for (int kk = 0; kk < 128; kk += 32) {
      v16h a0 = load_afrag(A, 1024, k0 + kk, l16, g);
      v16h a1 = load_afrag(A + (size_t)16 * 1024, 1024, k0 + kk, l16, g);
      __builtin_prefetch(A + (size_t)l16 * 1024 + k0 + kk + 256, 0, 1);
      v16h bf[4];
#pragma unroll
      for (int t = 0; t < 4; ++t)
        bf[t] = *(const v16h*)&wp[(size_t)(t * 16 + l16) * 128 + kk + g * 16];
#pragma unroll
      for (int t = 0; t < 4; ++t) acc[0][t] = wmma16(a0, bf[t], acc[0][t]);
#pragma unroll
      for (int t = 0; t < 4; ++t) acc[1][t] = wmma16(a1, bf[t], acc[1][t]);
    }
    __syncthreads();                    // panel c fully consumed before overwrite
  }
  // C/D layout: row M = g*8 + r (vgpr r), col N = l16 (+16*t)
  const int b = m0 >> 11;
  const int bh = b * 16 + h;
#pragma unroll
  for (int u = 0; u < 2; ++u) {
    const int sbase = (m0 & 2047) + u * 16 + g * 8;
#pragma unroll
    for (int t = 0; t < 4; ++t) {
      const int e = t * 16 + l16;
#pragma unroll
      for (int r = 0; r < 8; ++r) {
        _Float16 v = (_Float16)acc[u][t][r];
        const int s = sbase + r;
        if (which == 2)      Vt[((size_t)bh * 64 + e) * 2048 + s] = v;   // V transposed
        else if (which == 1) Kh[((size_t)bh * 2048 + s) * 64 + e] = v;
        else                 Q [((size_t)bh * 2048 + s) * 64 + e] = v;   // wq pre-scaled 1/8
      }
    }
  }
}

// ---------------- flash attention: one wave per (b,h,16-query tile) ----------------

__global__ __launch_bounds__(256) void k_attn(const _Float16* __restrict__ Q,
                                              const _Float16* __restrict__ Kh,
                                              const _Float16* __restrict__ Vt,
                                              const unsigned char* __restrict__ pad,
                                              _Float16* __restrict__ Hc) {
  __shared__ _Float16 plds[8][512];   // per-wave 16x32 f16 P tile
  const int wid = threadIdx.x >> 5, lane = threadIdx.x & 31;
  const int l16 = lane & 15, g = lane >> 4;
  const int id = blockIdx.x * 8 + wid;            // 8192 wave tiles
  const int qt = id & 127, h = (id >> 7) & 15, b = id >> 11;
  const int bh = b * 16 + h, q0 = qt * 16;

  const _Float16* qp = Q + ((size_t)bh * 2048 + q0 + l16) * 64 + g * 16;
  const v16h aq0 = cat8(*(const v8h*)qp, *(const v8h*)(qp + 8));
  const v16h aq1 = cat8(*(const v8h*)(qp + 32), *(const v8h*)(qp + 40));

  v8f acc[4] = {};
  float mrow[8], lrow[8];
#pragma unroll
  for (int r = 0; r < 8; ++r) { mrow[r] = -__builtin_inff(); lrow[r] = 0.f; }
  const int qbase = q0 + g * 8;

  for (int j0 = 0; j0 <= q0 + 15; j0 += 32) {
    // --- S = Q K^T (two 16-key tiles, Dh=64 -> 2 WMMA each) ---
    const _Float16* kr0 = Kh + ((size_t)bh * 2048 + j0 + l16) * 64 + g * 16;
    const _Float16* kr1 = kr0 + (size_t)16 * 64;
    v8f s0 = {}, s1 = {};
    s0 = wmma16(aq0, *(const v16h*)kr0,        s0);
    s0 = wmma16(aq1, *(const v16h*)(kr0 + 32), s0);
    s1 = wmma16(aq0, *(const v16h*)kr1,        s1);
    s1 = wmma16(aq1, *(const v16h*)(kr1 + 32), s1);

    // --- causal + key-pad mask (row = qbase+r, key = t0/t1) ---
    const int t0 = j0 + l16, t1 = t0 + 16;
    const bool ok0 = pad[(size_t)b * 2048 + t0] != 0;
    const bool ok1 = pad[(size_t)b * 2048 + t1] != 0;
#pragma unroll
    for (int r = 0; r < 8; ++r) {
      const int q = qbase + r;
      if (!ok0 || t0 > q) s0[r] = -__builtin_inff();
      if (!ok1 || t1 > q) s1[r] = -__builtin_inff();
    }

    // --- online softmax: row reductions across the 16-lane half ---
    float rmax[8];
#pragma unroll
    for (int r = 0; r < 8; ++r) rmax[r] = fmaxf(s0[r], s1[r]);
#pragma unroll
    for (int sh = 1; sh < 16; sh <<= 1)
#pragma unroll
      for (int r = 0; r < 8; ++r) rmax[r] = fmaxf(rmax[r], __shfl_xor(rmax[r], sh, 32));

    float p0[8], p1[8], rsum[8], scl[8];
#pragma unroll
    for (int r = 0; r < 8; ++r) {
      const float mn = fmaxf(mrow[r], rmax[r]);
      scl[r]  = exp2f((mrow[r] - mn) * LOG2E);
      p0[r]   = exp2f((s0[r] - mn) * LOG2E);
      p1[r]   = exp2f((s1[r] - mn) * LOG2E);
      rsum[r] = p0[r] + p1[r];
      mrow[r] = mn;
    }
#pragma unroll
    for (int sh = 1; sh < 16; sh <<= 1)
#pragma unroll
      for (int r = 0; r < 8; ++r) rsum[r] += __shfl_xor(rsum[r], sh, 32);
#pragma unroll
    for (int r = 0; r < 8; ++r) lrow[r] = lrow[r] * scl[r] + rsum[r];
#pragma unroll
    for (int t = 0; t < 4; ++t)
#pragma unroll
      for (int r = 0; r < 8; ++r) acc[t][r] *= scl[r];

    // --- reshape P (C/D layout) -> A fragment via per-wave LDS tile ---
    _Float16* pt = &plds[wid][0];
#pragma unroll
    for (int r = 0; r < 8; ++r) {
      pt[(g * 8 + r) * 32 + l16]      = (_Float16)p0[r];
      pt[(g * 8 + r) * 32 + 16 + l16] = (_Float16)p1[r];
    }
    asm volatile("s_wait_dscnt 0" ::: "memory");   // same-wave LDS RAW
    const _Float16* ar = pt + l16 * 32 + g * 16;
    const v16h ap = cat8(*(const v8h*)ar, *(const v8h*)(ar + 8));
    asm volatile("" ::: "memory");

    // --- O += P V  (V stored transposed [b,h,e,t]) ---
#pragma unroll
    for (int t = 0; t < 4; ++t) {
      const _Float16* vp = Vt + ((size_t)bh * 64 + t * 16 + l16) * 2048 + j0 + g * 16;
      acc[t] = wmma16(ap, *(const v16h*)vp, acc[t]);
    }
  }

  float inv[8];
#pragma unroll
  for (int r = 0; r < 8; ++r) inv[r] = 1.f / lrow[r];
#pragma unroll
  for (int t = 0; t < 4; ++t) {
    const int e = h * 64 + t * 16 + l16;
#pragma unroll
    for (int r = 0; r < 8; ++r)
      Hc[((size_t)b * 2048 + qbase + r) * 1024 + e] = (_Float16)(acc[t][r] * inv[r]);
  }
}

// ---------------- output projection, 32x64 wave tile, double-buffered TDM ----------------
// blockIdx.x & 15 -> 64-col weight panel; (blockIdx.x >> 4)*8 + wave -> 32-row tile.

__global__ __launch_bounds__(256) void k_out(const _Float16* __restrict__ Hc,
                                             const _Float16* __restrict__ woT,
                                             float* __restrict__ out) {
  __shared__ _Float16 wlds[2][64 * 128];   // 2 x 16 KB weight panels
  const int wid = threadIdx.x >> 5, lane = threadIdx.x & 31;
  const int l16 = lane & 15, g = lane >> 4;
  const int n0 = (blockIdx.x & 15) * 64;
  const int m0 = (((blockIdx.x >> 4) * 8) + wid) * 32;
  const _Float16* A = Hc + (size_t)m0 * 1024;
  const _Float16* Bt = woT + (size_t)n0 * 1024;
  const unsigned lds0 = (unsigned)(size_t)(void*)&wlds[0][0];
  const unsigned lds1 = (unsigned)(size_t)(void*)&wlds[1][0];

  v8f acc[2][4] = {};
  if (wid == 0)
    tdm_load_2d_f16(Bt, lds0, 128, 64, 1024, 64, 1024);
  for (int c = 0; c < 8; ++c) {
    if (wid == 0) {
      if (c + 1 < 8) {
        tdm_load_2d_f16(Bt + (c + 1) * 128, ((c + 1) & 1) ? lds1 : lds0,
                        128, 64, 1024, 64, 1024);
        __builtin_amdgcn_s_wait_tensorcnt(1);
      } else {
        __builtin_amdgcn_s_wait_tensorcnt(0);
      }
    }
    __syncthreads();
    const _Float16* wp = &wlds[c & 1][0];
    const int k0 = c * 128;
#pragma unroll
    for (int kk = 0; kk < 128; kk += 32) {
      v16h a0 = load_afrag(A, 1024, k0 + kk, l16, g);
      v16h a1 = load_afrag(A + (size_t)16 * 1024, 1024, k0 + kk, l16, g);
      __builtin_prefetch(A + (size_t)l16 * 1024 + k0 + kk + 256, 0, 1);
      v16h bf[4];
#pragma unroll
      for (int t = 0; t < 4; ++t)
        bf[t] = *(const v16h*)&wp[(size_t)(t * 16 + l16) * 128 + kk + g * 16];
#pragma unroll
      for (int t = 0; t < 4; ++t) acc[0][t] = wmma16(a0, bf[t], acc[0][t]);
#pragma unroll
      for (int t = 0; t < 4; ++t) acc[1][t] = wmma16(a1, bf[t], acc[1][t]);
    }
    __syncthreads();
  }
#pragma unroll
  for (int u = 0; u < 2; ++u)
#pragma unroll
    for (int t = 0; t < 4; ++t)
#pragma unroll
      for (int r = 0; r < 8; ++r)
        out[(size_t)(m0 + u * 16 + g * 8 + r) * 1024 + n0 + t * 16 + l16] = acc[u][t][r];
}

// ---------------- launch ----------------

extern "C" void kernel_launch(void* const* d_in, const int* in_sizes, int n_in,
                              void* d_out, int out_size, void* d_ws, size_t ws_size,
                              hipStream_t stream) {
  const float* embed = (const float*)d_in[0];
  const unsigned char* pad = (const unsigned char*)d_in[1];  // jnp bool -> 1 byte
  const float* w_q = (const float*)d_in[2];
  const float* w_k = (const float*)d_in[3];
  const float* w_v = (const float*)d_in[4];
  const float* w_o = (const float*)d_in[5];
  float* out = (float*)d_out;
  (void)in_sizes; (void)n_in; (void)out_size; (void)ws_size;

  char* ws = (char*)d_ws;
  size_t off = 0;
  auto take = [&](size_t elems) {
    _Float16* p = (_Float16*)(ws + off);
    off = (off + elems * 2 + 255) & ~(size_t)255;
    return p;
  };
  _Float16* embedH = take(8192ull * 1024);   // f16 embed
  _Float16* wqT    = take(1048576);          // [16][64][1024], pre-scaled 1/sqrt(64)
  _Float16* wkT    = take(1048576);
  _Float16* wvT    = take(1048576);
  _Float16* woT    = take(1048576);          // [1024][1024] transposed
  _Float16* Qh     = take(8388608);          // [b,h,s,e]
  _Float16* Kh     = take(8388608);          // [b,h,s,e]
  _Float16* Vt     = take(8388608);          // [b,h,e,s]
  _Float16* Hc     = take(8388608);          // heads_cat f16 [b,s,h*64+e]

  k_cvt_f16<<<8388608 / 256, 256, 0, stream>>>(embed, embedH, 8388608);
  k_cvt_wT <<<1048576 / 256, 256, 0, stream>>>(w_q, wqT, 0.125f);
  k_cvt_wT <<<1048576 / 256, 256, 0, stream>>>(w_k, wkT, 1.0f);
  k_cvt_wT <<<1048576 / 256, 256, 0, stream>>>(w_v, wvT, 1.0f);
  k_cvt_woT<<<1048576 / 256, 256, 0, stream>>>(w_o, woT);
  k_qkv <<<dim3(32, 48), 256, 0, stream>>>(embedH, wqT, wkT, wvT, Qh, Kh, Vt);
  k_attn<<<1024, 256, 0, stream>>>(Qh, Kh, Vt, pad, Hc);
  k_out <<<512, 256, 0, stream>>>(Hc, woT, out);
}
